// Posit_Drmm_Modeler_29669634081034
// MI455X (gfx1250) — matrix operand order repaired and checked by reading, not compiled, not run on hardware
//
#include <hip/hip_runtime.h>
#include <hip/hip_bf16.h>
#include <math.h>

// ---------------- problem constants ----------------
constexpr int B_  = 32;
constexpr int S_  = 32;
constexpr int LQ  = 30;
constexpr int LS  = 40;
constexpr int D_  = 300;
constexpr int DP  = 320;   // D padded to 10 * 32 (WMMA bf16 K step)
constexpr int F_  = 256;
constexpr int FS  = 3;

// ---------------- vector types ----------------
typedef __attribute__((ext_vector_type(16))) __bf16          v16bf;
typedef __attribute__((ext_vector_type(16))) unsigned short  v16u;
typedef __attribute__((ext_vector_type(8)))  float           v8f;

static __device__ inline unsigned short f2bf(float f) {
    unsigned int u = __float_as_uint(f);
    u += 0x7FFFu + ((u >> 16) & 1u);          // round-to-nearest-even
    return (unsigned short)(u >> 16);
}

// Swap bits 3 and 4 of k (involution). Storing each 32-K block of an
// A-matrix with this permutation makes a lane's A fragment (K order
// 16h..16h+15) land on two 16-byte chunks at +16*h and +32+16*h bytes.
static __device__ inline int perm32(int k) {
    return (k & ~24) | ((k & 8) << 1) | ((k & 16) >> 1);
}

static __device__ inline v16u zero16u() {
    v16u z;
#pragma unroll
    for (int i = 0; i < 16; ++i) z[i] = 0;
    return z;
}
static __device__ inline v8f zero8f() {
    v8f z;
#pragma unroll
    for (int i = 0; i < 8; ++i) z[i] = 0.0f;
    return z;
}

// D = A(16x32 bf16) x B(32x16 bf16) + C(16x16 f32), wave32
static __device__ inline v8f wmma_bf16(v16u a, v16u b, v8f c) {
    return __builtin_amdgcn_wmma_f32_16x16x32_bf16(
        /*neg_a=*/false, __builtin_bit_cast(v16bf, a),
        /*neg_b=*/false, __builtin_bit_cast(v16bf, b),
        /*c_mod=*/(short)0, c, /*reuse_a=*/false, /*reuse_b=*/false);
}

// A fragment from a bit-3/4-permuted buffer: two 16-byte chunks per lane (element i -> K = 16h + i).
static __device__ inline v16u loadAp(const unsigned short* rowBase, int kbase, int h) {
    typedef __attribute__((ext_vector_type(8))) unsigned short v8u;
    v8u lo = *reinterpret_cast<const v8u*>(rowBase + kbase + h * 8);
    v8u hi = *reinterpret_cast<const v8u*>(rowBase + kbase + 16 + h * 8);
    return __builtin_shufflevector(lo, hi, 0, 1, 2, 3, 4, 5, 6, 7, 8, 9, 10, 11, 12, 13, 14, 15);
}
// B fragment (natural order): lane l holds column n = l&15; element i -> K = 16h + i.
static __device__ inline v16u loadB(const unsigned short* p) {
    return *reinterpret_cast<const v16u*>(p);
}

// ======================================================================
// 1) Embedding gathers -> bf16 + f32 row norms
//    qe_bf is stored K-PERMUTED (A-only matrix); se_bf natural (B role too).
// ======================================================================
__global__ __launch_bounds__(256) void gather_q_kernel(
    const int* __restrict__ question, const float* __restrict__ embeds,
    unsigned short* __restrict__ qe_bf, float* __restrict__ qn) {
    const int row  = blockIdx.x * 8 + (threadIdx.x >> 5);  // 1024 rows (B*32)
    const int lane = threadIdx.x & 31;
    const int b = row >> 5, t = row & 31;
    unsigned short* dst = qe_bf + (size_t)row * DP;
    if (t >= LQ) {                       // padded rows 30,31 -> zeros
#pragma unroll
        for (int j = 0; j < 10; ++j) dst[lane + 32 * j] = 0;
        return;
    }
    const int tok = question[b * LQ + t];
    const float* e = embeds + (size_t)tok * D_;
    const int pl = perm32(lane);
    float ss = 0.0f;
#pragma unroll
    for (int j = 0; j < 10; ++j) {
        int d = lane + 32 * j;
        float v = (d < D_) ? e[d] : 0.0f;
        ss += v * v;
        dst[32 * j + pl] = f2bf(v);      // permuted store
    }
#pragma unroll
    for (int off = 16; off > 0; off >>= 1) ss += __shfl_down(ss, off);
    if (lane == 0) qn[b * LQ + t] = sqrtf(ss);
}

__global__ __launch_bounds__(256) void gather_s_kernel(
    const int* __restrict__ sentences, const float* __restrict__ embeds,
    unsigned short* __restrict__ se_bf, float* __restrict__ sn) {
    const int row  = blockIdx.x * 8 + (threadIdx.x >> 5);  // 40960 rows
    const int lane = threadIdx.x & 31;
    const int tok = sentences[row];
    const float* e = embeds + (size_t)tok * D_;
    unsigned short* dst = se_bf + (size_t)row * DP;
    float ss = 0.0f;
#pragma unroll
    for (int j = 0; j < 10; ++j) {
        int d = lane + 32 * j;
        float v = (d < D_) ? e[d] : 0.0f;
        ss += v * v;
        dst[d] = f2bf(v);                // natural order (B-matrix role)
    }
#pragma unroll
    for (int off = 16; off > 0; off >>= 1) ss += __shfl_down(ss, off);
    if (lane == 0) sn[row] = sqrtf(ss);
}

// ======================================================================
// 2) Pack conv filters into B-fragment order:
//    Wpack[((k*10+ks)*16 + ntile)*512 + lane*16 + i]
// ======================================================================
__global__ __launch_bounds__(32) void pack_w_kernel(
    const float* __restrict__ filters, unsigned short* __restrict__ Wpack) {
    const int frag = blockIdx.x;          // 3*10*16 = 480 fragments
    const int lane = threadIdx.x;
    const int nt = frag & 15;
    const int ks = (frag >> 4) % 10;
    const int k  = frag / 160;
    const int f  = nt * 16 + (lane & 15);
    const int h  = lane >> 4;
    unsigned short* dst = Wpack + (size_t)frag * 512 + lane * 16;
#pragma unroll
    for (int i = 0; i < 16; ++i) {
        int d = ks * 32 + h * 16 + i;
        float v = (d < D_) ? filters[(size_t)f * (FS * D_) + k * D_ + d] : 0.0f;
        dst[i] = f2bf(v);
    }
}

// ======================================================================
// 3) Conv-as-GEMM, sentences: one block per (b,s). out[o,f] = sum_k x[o+k]·W_k
//    M pad 48, N=256.  LDS staged K-PERMUTED; 2 N-tiles share each A fragment.
// ======================================================================
__global__ __launch_bounds__(256) void conv_s_kernel(
    const unsigned short* __restrict__ se_bf,
    const unsigned short* __restrict__ Wpack,
    unsigned short* __restrict__ sc_bf, float* __restrict__ scn) {
    __shared__ alignas(32) unsigned short X[50 * DP];   // rows >= 40 zero
    __shared__ float norms[48];
    const int bs = blockIdx.x;              // 0..1023
    const int tid = threadIdx.x, lane = tid & 31, wv = tid >> 5;
    const unsigned short* src = se_bf + (size_t)bs * LS * DP;
    for (int idx = tid; idx < 50 * DP; idx += 256) {
        int r = idx / DP, d = idx % DP;
        int ds = (d & ~31) + perm32(d & 31);            // permuted gather
        X[idx] = (r < LS) ? src[(size_t)r * DP + ds] : (unsigned short)0;
    }
    if (tid < 48) norms[tid] = 0.0f;
    __syncthreads();

    const int h = lane >> 4;
    for (int task = wv; task < 24; task += 8) {   // 3 Mtiles x 8 Ntile-pairs
        const int mt = task / 8, ntp = task & 7;  // ntiles ntp and ntp+8
        const int rowA = mt * 16 + (lane & 15);
        v8f acc0 = zero8f(), acc1 = zero8f();
#pragma unroll
        for (int k = 0; k < FS; ++k) {
#pragma unroll
            for (int ks = 0; ks < 10; ++ks) {
                v16u a  = loadAp(X + (rowA + k) * DP, ks * 32, h);
                const unsigned short* wb =
                    Wpack + (size_t)((k * 10 + ks) * 16) * 512 + lane * 16;
                v16u b0 = loadB(wb + (size_t)ntp * 512);
                v16u b1 = loadB(wb + (size_t)(ntp + 8) * 512);
                acc0 = wmma_bf16(a, b0, acc0);
                acc1 = wmma_bf16(a, b1, acc1);
            }
        }
        const int n0 = ntp * 16 + (lane & 15);
#pragma unroll
        for (int r = 0; r < 8; ++r) {
            int o = mt * 16 + r + 8 * h;
            if (o < LS) {
                float v0 = acc0[r], v1 = acc1[r];
                size_t base = (size_t)bs * LS * F_ + (size_t)o * F_;
                sc_bf[base + n0]       = f2bf(v0);
                sc_bf[base + n0 + 128] = f2bf(v1);
                atomicAdd(&norms[o], v0 * v0 + v1 * v1);
            }
        }
    }
    __syncthreads();
    if (tid < LS) scn[bs * LS + tid] = sqrtf(norms[tid]);
}

// Conv for questions: one block per b; M pad 32, rows 30/31 come out zero.
// qe_bf is already K-permuted -> direct LDS copy; qc_bf written K-permuted
// (it is only ever used as an A matrix).
__global__ __launch_bounds__(256) void conv_q_kernel(
    const unsigned short* __restrict__ qe_bf,
    const unsigned short* __restrict__ Wpack,
    unsigned short* __restrict__ qc_bf, float* __restrict__ qcn) {
    __shared__ alignas(32) unsigned short X[34 * DP];
    __shared__ float norms[32];
    const int b = blockIdx.x;
    const int tid = threadIdx.x, lane = tid & 31, wv = tid >> 5;
    const unsigned short* src = qe_bf + (size_t)b * 32 * DP;
    for (int idx = tid; idx < 34 * DP; idx += 256) {
        int r = idx / DP;
        X[idx] = (r < 32) ? src[idx] : (unsigned short)0;   // already permuted
    }
    if (tid < 32) norms[tid] = 0.0f;
    __syncthreads();

    const int h = lane >> 4;
    for (int task = wv; task < 16; task += 8) {   // 2 Mtiles x 8 Ntile-pairs
        const int mt = task / 8, ntp = task & 7;
        const int rowA = mt * 16 + (lane & 15);
        v8f acc0 = zero8f(), acc1 = zero8f();
#pragma unroll
        for (int k = 0; k < FS; ++k) {
#pragma unroll
            for (int ks = 0; ks < 10; ++ks) {
                v16u a  = loadAp(X + (rowA + k) * DP, ks * 32, h);
                const unsigned short* wb =
                    Wpack + (size_t)((k * 10 + ks) * 16) * 512 + lane * 16;
                v16u b0 = loadB(wb + (size_t)ntp * 512);
                v16u b1 = loadB(wb + (size_t)(ntp + 8) * 512);
                acc0 = wmma_bf16(a, b0, acc0);
                acc1 = wmma_bf16(a, b1, acc1);
            }
        }
        const int n0 = ntp * 16 + (lane & 15);
        const int p0 = (n0 & ~31) + perm32(n0 & 31);          // permuted col
        const int p1 = ((n0 + 128) & ~31) + perm32(n0 & 31);
#pragma unroll
        for (int r = 0; r < 8; ++r) {
            int o = mt * 16 + r + 8 * h;
            float v0 = acc0[r], v1 = acc1[r];
            size_t base = (size_t)b * 32 * F_ + (size_t)o * F_;
            qc_bf[base + p0] = f2bf(v0);                       // o<32
            qc_bf[base + p1] = f2bf(v1);
            if (o < LQ) atomicAdd(&norms[o], v0 * v0 + v1 * v1);
        }
    }
    __syncthreads();
    if (tid < LQ) qcn[b * LQ + tid] = sqrtf(norms[tid]);
}

// ======================================================================
// 4) Similarities (2 WMMA GEMMs) + top-5 pooling + logistic + mean over LQ
// ======================================================================
__global__ __launch_bounds__(256) void sim_pool_kernel(
    const unsigned short* __restrict__ qe_bf, const float* __restrict__ qn,
    const unsigned short* __restrict__ se_bf, const float* __restrict__ sn,
    const unsigned short* __restrict__ qc_bf, const float* __restrict__ qcn,
    const unsigned short* __restrict__ sc_bf, const float* __restrict__ scn,
    const float* __restrict__ soh, const float* __restrict__ lin_w,
    const float* __restrict__ lin_b, float* __restrict__ outSent) {
    __shared__ float sim[2][32][48];
    __shared__ float feats[LQ][6];
    __shared__ float lo[LQ];
    const int bs = blockIdx.x;
    const int b = bs >> 5;
    const int tid = threadIdx.x, lane = tid & 31, wv = tid >> 5;
    const int h = lane >> 4;

    // phase 1: 12 tile tasks = {sim_ins, sim_sen} x 2 Mtiles x 3 Ntiles
    for (int task = wv; task < 12; task += 8) {
        const int mat = task / 6;
        const int t6 = task % 6;
        const int mt = t6 / 3, nt = t6 % 3;
        const int rowA = mt * 16 + (lane & 15);
        const int n = nt * 16 + (lane & 15);
        const bool nOk = (n < LS);
        v8f acc = zero8f();
        if (mat == 0) {   // instance cosine numerator over D (K=320)
            const unsigned short* A = qe_bf + (size_t)b * 32 * DP;   // permuted
            const unsigned short* Bm = se_bf + (size_t)bs * LS * DP; // natural
#pragma unroll
            for (int ks = 0; ks < 10; ++ks) {
                v16u a = loadAp(A + rowA * DP, ks * 32, h);
                v16u bf_ = nOk ? loadB(Bm + (size_t)n * DP + ks * 32 + h * 16)
                               : zero16u();
                acc = wmma_bf16(a, bf_, acc);
            }
        } else {          // conv-feature cosine numerator over F (K=256)
            const unsigned short* A = qc_bf + (size_t)b * 32 * F_;    // permuted
            const unsigned short* Bm = sc_bf + (size_t)bs * LS * F_;  // natural
#pragma unroll
            for (int ks = 0; ks < 8; ++ks) {
                v16u a = loadAp(A + rowA * F_, ks * 32, h);
                v16u bf_ = nOk ? loadB(Bm + (size_t)n * F_ + ks * 32 + h * 16)
                               : zero16u();
                acc = wmma_bf16(a, bf_, acc);
            }
        }
#pragma unroll
        for (int r = 0; r < 8; ++r) {
            int m = mt * 16 + r + 8 * h;
            float val = 0.0f;
            if (m < LQ && n < LS) {
                float qv = mat ? qcn[b * LQ + m] : qn[b * LQ + m];
                float sv = mat ? scn[bs * LS + n] : sn[bs * LS + n];
                val = acc[r] / (qv * sv);
            }
            sim[mat][m][n] = val;
        }
    }
    __syncthreads();

    // phase 2: top-5 pooling per query row
    if (tid < LQ || (tid >= 32 && tid < 32 + LQ)) {
        const int mat = (tid >= 32) ? 1 : 0;
        const int q = mat ? (tid - 32) : tid;
        float t5[5] = {-1e30f, -1e30f, -1e30f, -1e30f, -1e30f};
        for (int j = 0; j < LS; ++j) {
            float v = sim[mat][q][j];
            if (v > t5[4]) {
                t5[4] = v;
#pragma unroll
                for (int k = 4; k > 0; --k)
                    if (t5[k] > t5[k - 1]) { float t = t5[k]; t5[k] = t5[k - 1]; t5[k - 1] = t; }
            }
        }
        feats[q][2 * mat + 0] = t5[0];
        feats[q][2 * mat + 1] = 0.2f * (t5[0] + t5[1] + t5[2] + t5[3] + t5[4]);
    } else if (tid >= 64 && tid < 64 + LQ) {
        const int q = tid - 64;                 // one-hot pooling analytically
        const float* row = soh + ((size_t)bs * LQ + q) * LS;
        float cnt = 0.0f;
        for (int j = 0; j < LS; ++j) cnt += row[j];
        feats[q][4] = (cnt > 0.5f) ? 1.0f : 0.0f;
        feats[q][5] = fminf(cnt, 5.0f) * 0.2f;
    }
    __syncthreads();

    // phase 3: logistic per row, then mean over LQ
    if (tid < LQ) {
        float z = lin_b[0];
#pragma unroll
        for (int i = 0; i < 6; ++i) z += feats[tid][i] * lin_w[i];
        lo[tid] = 1.0f / (1.0f + expf(-z));
    }
    __syncthreads();
    if (tid == 0) {
        float s = 0.0f;
        for (int q = 0; q < LQ; ++q) s += lo[q];
        outSent[bs] = s / (float)LQ;
    }
}

// ======================================================================
// 5) Losses: doc_em = max over S, clipped BCEs, total loss
// ======================================================================
__global__ __launch_bounds__(32) void loss_kernel(
    const float* __restrict__ sent, const int* __restrict__ tS,
    const int* __restrict__ tD, float* __restrict__ d_out) {
    const int b = threadIdx.x;   // one wave, one doc per lane
    float bsum = 0.0f, pmax = -1e30f;
    for (int s = 0; s < S_; ++s) {
        float p = sent[b * S_ + s];
        float t = (float)tS[b * S_ + s];
        float lp = fmaxf(logf(p), -100.0f);
        float l1 = fmaxf(logf(1.0f - p), -100.0f);
        bsum += -(t * lp + (1.0f - t) * l1);
        pmax = fmaxf(pmax, p);
    }
    d_out[1 + B_ * S_ + b] = pmax;                      // doc_em
    float td = (float)tD[b];
    float dl = -(td * fmaxf(logf(pmax), -100.0f) +
                 (1.0f - td) * fmaxf(logf(1.0f - pmax), -100.0f));
    float s1 = bsum, s2 = dl;
#pragma unroll
    for (int off = 16; off > 0; off >>= 1) {
        s1 += __shfl_down(s1, off);
        s2 += __shfl_down(s2, off);
    }
    if (b == 0)
        d_out[0] = 0.5f * (s1 / (float)(B_ * S_) + s2 / (float)B_);
}

// ======================================================================
// launch
// ======================================================================
extern "C" void kernel_launch(void* const* d_in, const int* in_sizes, int n_in,
                              void* d_out, int out_size, void* d_ws, size_t ws_size,
                              hipStream_t stream) {
    (void)in_sizes; (void)n_in; (void)out_size; (void)ws_size;
    const int*   question  = (const int*)d_in[0];
    const int*   sentences = (const int*)d_in[1];
    const int*   tS        = (const int*)d_in[2];
    const int*   tD        = (const int*)d_in[3];
    const float* soh       = (const float*)d_in[4];
    const float* embeds    = (const float*)d_in[5];
    const float* filters   = (const float*)d_in[6];
    const float* lin_w     = (const float*)d_in[7];
    const float* lin_b     = (const float*)d_in[8];

    char* ws = (char*)d_ws;
    size_t off = 0;
    auto alloc = [&](size_t bytes) {
        size_t o = off;
        off += (bytes + 255) & ~(size_t)255;
        return o;
    };
    unsigned short* qe_bf = (unsigned short*)(ws + alloc((size_t)B_ * 32 * DP * 2));
    float*          qn    = (float*)(ws + alloc((size_t)B_ * LQ * 4));
    unsigned short* se_bf = (unsigned short*)(ws + alloc((size_t)B_ * S_ * LS * DP * 2));
    float*          sn    = (float*)(ws + alloc((size_t)B_ * S_ * LS * 4));
    unsigned short* Wpack = (unsigned short*)(ws + alloc((size_t)480 * 512 * 2));
    unsigned short* qc_bf = (unsigned short*)(ws + alloc((size_t)B_ * 32 * F_ * 2));
    float*          qcn   = (float*)(ws + alloc((size_t)B_ * LQ * 4));
    unsigned short* sc_bf = (unsigned short*)(ws + alloc((size_t)B_ * S_ * LS * F_ * 2));
    float*          scn   = (float*)(ws + alloc((size_t)B_ * S_ * LS * 4));

    float* out = (float*)d_out;          // [0]=loss, [1..1024]=sent, [1025..1056]=doc_em

    gather_q_kernel<<<(B_ * 32) / 8, 256, 0, stream>>>(question, embeds, qe_bf, qn);
    gather_s_kernel<<<(B_ * S_ * LS) / 8, 256, 0, stream>>>(sentences, embeds, se_bf, sn);
    pack_w_kernel<<<480, 32, 0, stream>>>(filters, Wpack);
    conv_s_kernel<<<B_ * S_, 256, 0, stream>>>(se_bf, Wpack, sc_bf, scn);
    conv_q_kernel<<<B_, 256, 0, stream>>>(qe_bf, Wpack, qc_bf, qcn);
    sim_pool_kernel<<<B_ * S_, 256, 0, stream>>>(qe_bf, qn, se_bf, sn, qc_bf, qcn,
                                                 sc_bf, scn, soh, lin_w, lin_b,
                                                 out + 1);
    loss_kernel<<<1, 32, 0, stream>>>(out + 1, tS, tD, out);
}
